// QuantizedViTAttention_48318382080288
// MI455X (gfx1250) — 16-bit and 8-bit matrix operand order repaired and checked by reading, compiled, not run on hardware
//
#include <hip/hip_runtime.h>

#define HDIM   1024
#define NHEADS 16
#define HEADD  64
#define BATCH  16
#define SEQ    577
#define MROWS  (BATCH * SEQ)
#define SPAD   640
#define QT     37

typedef __attribute__((ext_vector_type(16))) __bf16 v16bf;
typedef __attribute__((ext_vector_type(8)))  float  v8f;
typedef __attribute__((ext_vector_type(8)))  int    v8i;

typedef unsigned int  u32;
typedef unsigned char u8;
typedef signed char   i8;

__device__ __forceinline__ u32 pack_bf16(float lo, float hi) {
  u32 a = __float_as_uint(lo);
  u32 b = __float_as_uint(hi);
  a = (a + 0x7fffu + ((a >> 16) & 1u)) >> 16;
  b = (b + 0x7fffu + ((b >> 16) & 1u)) & 0xffff0000u;
  return b | (a & 0xffffu);
}

__global__ void init_hdr_kernel(int* hdr) {
  if (threadIdx.x < 64) hdr[threadIdx.x] = 0;
}

__global__ __launch_bounds__(256) void wscale_kernel(
    const float* __restrict__ wq, const float* __restrict__ wk,
    const float* __restrict__ wv, float* __restrict__ swall) {
  const int z = blockIdx.y;
  const float* w = (z == 0) ? wq : (z == 1) ? wk : wv;
  const int row  = blockIdx.x * 8 + (threadIdx.x >> 5);
  const int lane = threadIdx.x & 31;
  const float* wr = w + (size_t)row * HDIM;
  float m = 0.f;
  for (int k = lane; k < HDIM; k += 32) m = fmaxf(m, fabsf(wr[k]));
#pragma unroll
  for (int d = 1; d < 32; d <<= 1) m = fmaxf(m, __shfl_xor(m, d, 32));
  if (lane == 0) swall[z * HDIM + row] = m * (1.f / 127.f);
}

__global__ __launch_bounds__(256) void qkv_gemm_kernel(
    const float* __restrict__ x,
    const float* __restrict__ wq, const float* __restrict__ wk, const float* __restrict__ wv,
    const float* __restrict__ bq, const float* __restrict__ bk, const float* __restrict__ bv,
    const float* __restrict__ swall, float* __restrict__ ybase, int* __restrict__ hdr) {
  const int z = blockIdx.z;
  const float* w    = (z == 0) ? wq : (z == 1) ? wk : wv;
  const float* bias = (z == 0) ? bq : (z == 1) ? bk : bv;
  const float* sw   = swall + z * HDIM;
  float* y = ybase + (size_t)z * MROWS * HDIM;

  const int tid  = threadIdx.x;
  const int wave = tid >> 5;
  const int lane = tid & 31;
  const int h    = lane >> 4;
  const int nl   = lane & 15;

  const int m0 = blockIdx.x * 128 + wave * 16;
  const int n0 = blockIdx.y * 64;

  int arow = m0 + nl;
  if (arow > MROWS - 1) arow = MROWS - 1;
  const float* xrow = x + (size_t)arow * HDIM;

  __shared__ u32 Bt[64 * 16];

  v8f acc[4];
#pragma unroll
  for (int nt = 0; nt < 4; ++nt) acc[nt] = (v8f){0.f,0.f,0.f,0.f,0.f,0.f,0.f,0.f};

  const int   sn   = tid >> 2;
  const int   sk   = (tid & 3) * 8;
  const float ssc  = sw[n0 + sn];
  const float sinv = (ssc > 0.f) ? (1.f / ssc) : 0.f;
  const float* wrow = w + (size_t)(n0 + sn) * HDIM + sk;

  for (int k0 = 0; k0 < HDIM; k0 += 32) {
    __syncthreads();
    {
      float4 w0 = *(const float4*)(wrow + k0);
      float4 w1 = *(const float4*)(wrow + k0 + 4);
      float f[8] = {w0.x, w0.y, w0.z, w0.w, w1.x, w1.y, w1.z, w1.w};
#pragma unroll
      for (int j = 0; j < 8; ++j) {
        float qv = fminf(fmaxf(rintf(f[j] * sinv), -128.f), 127.f) * ssc;
        f[j] = qv;
      }
      u32* dst = &Bt[sn * 16 + (sk >> 1)];
      dst[0] = pack_bf16(f[0], f[1]);
      dst[1] = pack_bf16(f[2], f[3]);
      dst[2] = pack_bf16(f[4], f[5]);
      dst[3] = pack_bf16(f[6], f[7]);
    }
    __syncthreads();

    union { u32 u[8]; v16bf v; } A;
    {
      const float4* xp  = (const float4*)(xrow + k0 + 16 * h);
      const float4* xp2 = (const float4*)(xrow + k0 + 16 * h + 8);
      float4 a0 = xp[0],  a1 = xp[1];
      float4 a2 = xp2[0], a3 = xp2[1];
      A.u[0] = pack_bf16(a0.x, a0.y); A.u[1] = pack_bf16(a0.z, a0.w);
      A.u[2] = pack_bf16(a1.x, a1.y); A.u[3] = pack_bf16(a1.z, a1.w);
      A.u[4] = pack_bf16(a2.x, a2.y); A.u[5] = pack_bf16(a2.z, a2.w);
      A.u[6] = pack_bf16(a3.x, a3.y); A.u[7] = pack_bf16(a3.z, a3.w);
    }
#pragma unroll
    for (int nt = 0; nt < 4; ++nt) {
      union { u32 u[8]; v16bf v; } Bf;
      const u32* bp = &Bt[(nt * 16 + nl) * 16 + h * 8];
#pragma unroll
      for (int j = 0; j < 8; ++j) Bf.u[j] = bp[j];
      acc[nt] = __builtin_amdgcn_wmma_f32_16x16x32_bf16(
          false, A.v, false, Bf.v, (short)0, acc[nt], false, false);
    }
  }

  float lmax = 0.f;
#pragma unroll
  for (int nt = 0; nt < 4; ++nt) {
    const int col = n0 + nt * 16 + nl;
    const float bvl = bias[col];
#pragma unroll
    for (int r = 0; r < 8; ++r) {
      float val = acc[nt][r] + bvl;
      int mg = m0 + r + 8 * h;
      if (mg < MROWS) y[(size_t)mg * HDIM + col] = val;
      lmax = fmaxf(lmax, fabsf(val));
    }
  }
#pragma unroll
  for (int d = 1; d < 32; d <<= 1) lmax = fmaxf(lmax, __shfl_xor(lmax, d, 32));
  if (lane == 0) atomicMax(&hdr[z], __float_as_int(lmax));
}

__device__ __forceinline__ i8 quant8(float v, float s) {
  if (s <= 0.f) return (i8)0;
  float r = fminf(fmaxf(rintf(v / s), -128.f), 127.f);
  return (i8)(int)r;
}

__global__ __launch_bounds__(256) void quant_qkv_kernel(
    const float* __restrict__ ybase, const int* __restrict__ hdr,
    i8* __restrict__ q8, i8* __restrict__ k8, i8* __restrict__ v8t) {
  const size_t idx = (size_t)blockIdx.x * 256 + threadIdx.x;
  const int d  = (int)(idx & 63);
  const int s  = (int)((idx >> 6) % SPAD);
  const int bh = (int)(idx / ((size_t)SPAD * 64));
  const int b = bh >> 4, hh = bh & 15;
  const float sq = __int_as_float(hdr[0]) * (1.f / 127.f);
  const float sk = __int_as_float(hdr[1]) * (1.f / 127.f);
  const float sv = __int_as_float(hdr[2]) * (1.f / 127.f);
  float qv = 0.f, kv = 0.f, vv = 0.f;
  if (s < SEQ) {
    size_t src = ((size_t)(b * SEQ + s)) * HDIM + hh * 64 + d;
    qv = ybase[src];
    kv = ybase[(size_t)MROWS * HDIM + src];
    vv = ybase[2ull * MROWS * HDIM + src];
  }
  q8[idx] = quant8(qv, sq);
  k8[idx] = quant8(kv, sk);
  v8t[((size_t)bh * 64 + d) * SPAD + s] = quant8(vv, sv);
}

__device__ __forceinline__ v8i qk_tile(const i8* __restrict__ kb, int kt, int h, int nl, v8i A) {
  const i8* krow = kb + (size_t)(kt * 16 + nl) * 64;
  uint4 b0 = *(const uint4*)(krow + 16 * h);
  uint4 b1 = *(const uint4*)(krow + 32 + 16 * h);
  union { u32 u[8]; v8i v; } Bf;
  Bf.u[0] = b0.x; Bf.u[1] = b0.y; Bf.u[2] = b0.z; Bf.u[3] = b0.w;
  Bf.u[4] = b1.x; Bf.u[5] = b1.y; Bf.u[6] = b1.z; Bf.u[7] = b1.w;
  v8i c = {0, 0, 0, 0, 0, 0, 0, 0};
  return __builtin_amdgcn_wmma_i32_16x16x64_iu8(true, A, true, Bf.v, c, false, false);
}

__device__ __forceinline__ void load_qfrag(const i8* __restrict__ qb, int qt, int h, int nl, u32* u) {
  const i8* qrow = qb + (size_t)(qt * 16 + nl) * 64;
  uint2 c0 = *(const uint2*)(qrow + 16 * h);
  uint2 c1 = *(const uint2*)(qrow + 8 + 16 * h);
  uint2 c2 = *(const uint2*)(qrow + 32 + 16 * h);
  uint2 c3 = *(const uint2*)(qrow + 40 + 16 * h);
  u[0] = c0.x; u[1] = c0.y; u[2] = c1.x; u[3] = c1.y;
  u[4] = c2.x; u[5] = c2.y; u[6] = c3.x; u[7] = c3.y;
}

__global__ __launch_bounds__(256) void attn_absmax_kernel(
    const i8* __restrict__ q8, const i8* __restrict__ k8, int* __restrict__ hdr) {
  const int bh = blockIdx.x;
  const int wave = threadIdx.x >> 5;
  const int lane = threadIdx.x & 31;
  const int qt = blockIdx.y * 8 + wave;
  if (qt >= QT) return;
  const int h = lane >> 4, nl = lane & 15;
  const i8* qb = q8 + (size_t)bh * SPAD * 64;
  const i8* kb = k8 + (size_t)bh * SPAD * 64;
  union { u32 u[8]; v8i v; } A;
  load_qfrag(qb, qt, h, nl, A.u);
  int lmax = 0;
  for (int kt = 0; kt < QT; ++kt) {
    v8i acc = qk_tile(kb, kt, h, nl, A.v);
#pragma unroll
    for (int r = 0; r < 8; ++r) {
      int a = acc[r]; a = (a < 0) ? -a : a;
      lmax = (lmax > a) ? lmax : a;
    }
  }
#pragma unroll
  for (int d = 1; d < 32; d <<= 1) {
    int o = __shfl_xor(lmax, d, 32);
    lmax = (lmax > o) ? lmax : o;
  }
  if (lane == 0) atomicMax(&hdr[3], lmax);
}

__device__ __forceinline__ float ibert_exp16(float xd, float x0i, float invx0,
                                             float bi, float ci, float ek) {
  float xi = fmaxf(xd, 30.f * x0i);
  float qf = floorf(xi * invx0);
  float rf = xi - x0i * qf;
  float z  = rf * (rf + bi) + ci;
  float e  = floorf(ldexpf(z, 30 - (int)qf));
  e = fmaxf(e, 0.f);
  return fminf(rintf(e * ek), 32767.f);
}

__global__ __launch_bounds__(128) void attn_main_kernel(
    const i8* __restrict__ q8, const i8* __restrict__ k8, const i8* __restrict__ v8t,
    int* __restrict__ hdr, float* __restrict__ out) {
  __shared__ u8 ptile[4 * 1024];
  const int bh = blockIdx.x;
  const int wave = threadIdx.x >> 5;
  const int lane = threadIdx.x & 31;
  const int qt = blockIdx.y * 4 + wave;
  if (qt >= QT) return;
  const int h = lane >> 4, nl = lane & 15;
  const int b = bh >> 4, head = bh & 15;

  const float sq = __int_as_float(hdr[0]) * (1.f / 127.f);
  const float sk = __int_as_float(hdr[1]) * (1.f / 127.f);
  const float sv = __int_as_float(hdr[2]) * (1.f / 127.f);
  const float s_attn = sq * sk * 0.125f;
  float maxacc = (float)hdr[3]; if (maxacc < 1.f) maxacc = 1.f;
  const float s     = fmaxf(maxacc * s_attn * (1.f / 32768.f), 1e-30f);
  const float ratio = 32768.f / maxacc;
  const float x0i   = floorf(-0.6931f / s);
  const float bi    = floorf((0.96963238f / 0.35815147f) / s);
  const float ci    = fmaxf(floorf((1.f / 0.35815147f) / (s * s)), 1.f);
  const float ek    = 32767.f / (ci * 1073741824.f);
  const float invx0 = 1.f / x0i;

  const i8* qb = q8  + (size_t)bh * SPAD * 64;
  const i8* kb = k8  + (size_t)bh * SPAD * 64;
  const i8* vb = v8t + (size_t)bh * 64 * SPAD;

  union { u32 u[8]; v8i v; } A;
  load_qfrag(qb, qt, h, nl, A.u);

  float xmax[8];
#pragma unroll
  for (int r = 0; r < 8; ++r) xmax[r] = -3.0e38f;
  for (int kt = 0; kt < QT; ++kt) {
    v8i acc = qk_tile(kb, kt, h, nl, A.v);
    const bool valid = (kt * 16 + nl) < SEQ;
#pragma unroll
    for (int r = 0; r < 8; ++r) {
      float xi = rintf((float)acc[r] * ratio);
      if (valid) xmax[r] = fmaxf(xmax[r], xi);
    }
  }
#pragma unroll
  for (int r = 0; r < 8; ++r) {
#pragma unroll
    for (int d = 1; d < 16; d <<= 1) xmax[r] = fmaxf(xmax[r], __shfl_xor(xmax[r], d, 32));
  }

  float rsum[8];
#pragma unroll
  for (int r = 0; r < 8; ++r) rsum[r] = 0.f;
  for (int kt = 0; kt < QT; ++kt) {
    v8i acc = qk_tile(kb, kt, h, nl, A.v);
    const bool valid = (kt * 16 + nl) < SEQ;
#pragma unroll
    for (int r = 0; r < 8; ++r) {
      if (valid) {
        float xd = rintf((float)acc[r] * ratio) - xmax[r];
        rsum[r] += ibert_exp16(xd, x0i, invx0, bi, ci, ek);
      }
    }
  }
  float fac[8];
#pragma unroll
  for (int r = 0; r < 8; ++r) {
#pragma unroll
    for (int d = 1; d < 16; d <<= 1) rsum[r] += __shfl_xor(rsum[r], d, 32);
    fac[r] = floorf(4294967296.f / fmaxf(rsum[r], 1.f));
  }

  v8i ctx[4];
#pragma unroll
  for (int nt = 0; nt < 4; ++nt) ctx[nt] = (v8i){0, 0, 0, 0, 0, 0, 0, 0};
  u8* tb = ptile + wave * 1024;

  for (int c = 0; c < SPAD / 64; ++c) {
    asm volatile("" ::: "memory");
#pragma unroll
    for (int i = 0; i < 4; ++i) {
      const int kt = c * 4 + i;
      v8i acc = qk_tile(kb, kt, h, nl, A.v);
      const bool valid = (kt * 16 + nl) < SEQ;
#pragma unroll
      for (int r = 0; r < 8; ++r) {
        float p = 0.f;
        if (valid) {
          float xd = rintf((float)acc[r] * ratio) - xmax[r];
          float e  = ibert_exp16(xd, x0i, invx0, bi, ci, ek);
          p = fminf(floorf(e * fac[r] * 5.9604644775390625e-8f), 255.f);
        }
        tb[(r + 8 * h) * 64 + i * 16 + nl] = (u8)(int)p;
      }
    }
    asm volatile("s_wait_dscnt 0x0" ::: "memory");

    union { u32 u[8]; v8i v; } P;
    const u8* pr = tb + nl * 64;
    uint2 p0 = *(const uint2*)(pr + 16 * h);
    uint2 p1 = *(const uint2*)(pr + 8 + 16 * h);
    uint2 p2 = *(const uint2*)(pr + 32 + 16 * h);
    uint2 p3 = *(const uint2*)(pr + 40 + 16 * h);
    P.u[0] = p0.x; P.u[1] = p0.y; P.u[2] = p1.x; P.u[3] = p1.y;
    P.u[4] = p2.x; P.u[5] = p2.y; P.u[6] = p3.x; P.u[7] = p3.y;

#pragma unroll
    for (int nt = 0; nt < 4; ++nt) {
      const i8* vrow = vb + (size_t)(nt * 16 + nl) * SPAD + c * 64;
      uint4 b0 = *(const uint4*)(vrow + 16 * h);
      uint4 b1 = *(const uint4*)(vrow + 32 + 16 * h);
      union { u32 u[8]; v8i v; } Bf;
      Bf.u[0] = b0.x; Bf.u[1] = b0.y; Bf.u[2] = b0.z; Bf.u[3] = b0.w;
      Bf.u[4] = b1.x; Bf.u[5] = b1.y; Bf.u[6] = b1.z; Bf.u[7] = b1.w;
      ctx[nt] = __builtin_amdgcn_wmma_i32_16x16x64_iu8(
          false  , P.v, true  , Bf.v, ctx[nt], false, false);
    }
    asm volatile("s_wait_dscnt 0x0" ::: "memory");
  }

  const float cs = sv * (1.f / 256.f);
  float lmax = 0.f;
#pragma unroll
  for (int nt = 0; nt < 4; ++nt) {
#pragma unroll
    for (int r = 0; r < 8; ++r) {
      const int srow = qt * 16 + r + 8 * h;
      if (srow < SEQ) {
        float val = (float)ctx[nt][r] * cs;
        out[((size_t)(b * SEQ + srow)) * HDIM + head * 64 + nt * 16 + nl] = val;
        lmax = fmaxf(lmax, fabsf(val));
      }
    }
  }
#pragma unroll
  for (int d = 1; d < 32; d <<= 1) lmax = fmaxf(lmax, __shfl_xor(lmax, d, 32));
  if (lane == 0) atomicMax(&hdr[4], __float_as_int(lmax));
}

__global__ __launch_bounds__(256) void out_quant_kernel(float* __restrict__ out,
                                                        const int* __restrict__ hdr) {
  const size_t i = (size_t)blockIdx.x * 256 + threadIdx.x;
  const float so = __int_as_float(hdr[4]) * (1.f / 127.f);
  if (so > 0.f) {
    float v = out[i];
    out[i] = fminf(fmaxf(rintf(v / so), -128.f), 127.f) * so;
  }
}

extern "C" void kernel_launch(void* const* d_in, const int* in_sizes, int n_in,
                              void* d_out, int out_size, void* d_ws, size_t ws_size,
                              hipStream_t stream) {
  const float* x  = (const float*)d_in[0];
  const float* wq = (const float*)d_in[1];
  const float* bq = (const float*)d_in[2];
  const float* wk = (const float*)d_in[3];
  const float* bk = (const float*)d_in[4];
  const float* wv = (const float*)d_in[5];
  const float* bv = (const float*)d_in[6];
  float* out = (float*)d_out;

  char* wsb = (char*)d_ws;
  int*   hdr   = (int*)wsb;
  float* swall = (float*)(wsb + 256);
  size_t yoff  = 256 + 3ull * HDIM * 4;
  float* ybase = (float*)(wsb + yoff);
  size_t ybytes = 3ull * MROWS * HDIM * 4;
  i8* q8  = (i8*)(wsb + yoff + ybytes);
  size_t t8 = (size_t)BATCH * NHEADS * SPAD * 64;
  i8* k8  = q8 + t8;
  i8* v8t = k8 + t8;
  (void)in_sizes; (void)n_in; (void)out_size; (void)ws_size;

  init_hdr_kernel<<<dim3(1), dim3(64), 0, stream>>>(hdr);
  wscale_kernel<<<dim3(128, 3), dim3(256), 0, stream>>>(wq, wk, wv, swall);
  qkv_gemm_kernel<<<dim3((MROWS + 127) / 128, HDIM / 64, 3), dim3(256), 0, stream>>>(
      x, wq, wk, wv, bq, bk, bv, swall, ybase, hdr);
  quant_qkv_kernel<<<dim3((unsigned)(t8 / 256)), dim3(256), 0, stream>>>(
      ybase, hdr, q8, k8, v8t);
  attn_absmax_kernel<<<dim3(BATCH * NHEADS, (QT + 7) / 8), dim3(256), 0, stream>>>(q8, k8, hdr);
  attn_main_kernel<<<dim3(BATCH * NHEADS, (QT + 3) / 4), dim3(128), 0, stream>>>(
      q8, k8, v8t, hdr, out);
  out_quant_kernel<<<dim3((unsigned)(((size_t)MROWS * HDIM) / 256)), dim3(256), 0, stream>>>(
      out, hdr);
}
